// mulityHeadAttnGCN_14903536517912
// MI455X (gfx1250) — matrix operand order repaired and checked by reading, not compiled, not run on hardware
//
#include <hip/hip_runtime.h>
#include <hip/hip_fp16.h>

// B=256, S=128, D=1024, H=8, DH=128.  BS = B*S = 32768.
typedef __attribute__((ext_vector_type(16))) _Float16 v16h;
typedef __attribute__((ext_vector_type(8)))  _Float16 h8;
typedef __attribute__((ext_vector_type(4)))  _Float16 h4;
typedef __attribute__((ext_vector_type(8)))  float    v8f;

#define WMMA_F32_F16(a, b, c) \
  __builtin_amdgcn_wmma_f32_16x16x32_f16(false, (a), false, (b), (short)0, (c), false, false)

// Concatenate two 8-half LDS vector loads into one fragment (2x ds_load_b128).
__device__ inline v16h ld_frag2(const _Float16* p0, const _Float16* p1) {
  const h8 lo = *(const h8*)p0;
  const h8 hi = *(const h8*)p1;
  return __builtin_shufflevector(lo, hi, 0, 1, 2, 3, 4, 5, 6, 7,
                                         8, 9, 10, 11, 12, 13, 14, 15);
}

// A: 16x32 f16 row-major [M x K].  Lane layout (ISA 7.12.2): row M = lane&15,
// halfs 0..7 = K[koff..koff+7], halfs 8..15 = K[koff+16..koff+23], koff = 8*(lane>=16).
__device__ inline v16h frag_a_lds(const _Float16* p, int ld, int m0, int k0) {
  const int lane = threadIdx.x & 31;
  const int m = m0 + (lane & 15);
  const int koff = (lane >> 4) << 3;
  const _Float16* base = p + m * ld + k0 + koff;
  return ld_frag2(base, base + 16);
}

// B: 32x16 from a TRANSPOSED [N x K] tile: col N = lane&15, halfs 0..7 = K[kh..kh+7],
// halfs 8..15 = K[kh+16..kh+23], kh = 8*(lane>=16).  2x ds_load_b128.
__device__ inline v16h frag_bt_lds(const _Float16* p, int ld, int k0, int n0) {
  const int lane = threadIdx.x & 31;
  const int n = n0 + (lane & 15);
  const int kh = (lane >> 4) << 3;
  const _Float16* base = p + n * ld + k0 + kh;
  return ld_frag2(base, base + 16);
}

// ---- Kernel 1: Q/K/V projection  relu(code @ W + b) -> f16 [H, B*S, DH] ----
// Double-buffered software pipeline: global loads for slab k+1 are in flight
// while WMMAs run on slab k; one barrier per K-step.
__global__ __launch_bounds__(256)
void qkv_proj_kernel(const float* __restrict__ code,   // [32768, 1024]
                     const float* __restrict__ W,      // [H, D, DH] = [8,1024,128]
                     const float* __restrict__ bias,   // [1024]
                     _Float16* __restrict__ out)       // [H, 32768, 128]
{
  __shared__ __align__(16) _Float16 sA [2][128 * 32];  // A tiles, row-major M x K
  __shared__ __align__(16) _Float16 sBt[2][128 * 32];  // B tiles transposed, N x K
  const int mBase = blockIdx.x * 128;
  const int nBase = blockIdx.y * 128;
  const int tid  = threadIdx.x;
  const int wave = tid >> 5;
  const int lane = tid & 31;

  v8f zero = {};
  v8f acc[8];
#pragma unroll
  for (int j = 0; j < 8; ++j) acc[j] = zero;

  const float* Whead = W + (size_t)(nBase >> 7) * 131072;  // one head per n-tile block

  // prologue: stage K-slab 0 into buffer 0
#pragma unroll
  for (int u = 0; u < 4; ++u) {
    const int i = tid * 4 + u * 1024;
    {
      const int r = i >> 5, c = i & 31;
      const float4 x = *(const float4*)&code[(size_t)(mBase + r) * 1024 + c];
      const h4 y = {(_Float16)x.x, (_Float16)x.y, (_Float16)x.z, (_Float16)x.w};
      *(h4*)&sA[0][i] = y;
    }
    {
      const int r = i >> 7, c = i & 127;
      const float4 x = *(const float4*)&Whead[(size_t)r * 128 + c];
      sBt[0][(c + 0) * 32 + r] = (_Float16)x.x;
      sBt[0][(c + 1) * 32 + r] = (_Float16)x.y;
      sBt[0][(c + 2) * 32 + r] = (_Float16)x.z;
      sBt[0][(c + 3) * 32 + r] = (_Float16)x.w;
    }
  }
  __syncthreads();

  for (int kt = 0; kt < 32; ++kt) {
    float4 ax[4], bx[4];
    const bool hasNext = (kt + 1) < 32;
    const int kn = (kt + 1) * 32;
    if (hasNext) {                                    // issue next-slab loads
#pragma unroll
      for (int u = 0; u < 4; ++u) {
        const int i = tid * 4 + u * 1024;
        const int ra = i >> 5, ca = i & 31;
        ax[u] = *(const float4*)&code[(size_t)(mBase + ra) * 1024 + kn + ca];
        const int rb = i >> 7, cb = i & 127;
        bx[u] = *(const float4*)&Whead[(size_t)(kn + rb) * 128 + cb];
      }
    }
    // compute on current buffer while loads are in flight
    const _Float16* A  = sA [kt & 1];
    const _Float16* Bt = sBt[kt & 1];
    const v16h a = frag_a_lds(A, 32, wave * 16, 0);
#pragma unroll
    for (int j = 0; j < 8; ++j) {
      const v16h b = frag_bt_lds(Bt, 32, 0, j * 16);
      acc[j] = WMMA_F32_F16(a, b, acc[j]);
    }
    // drain loads into the other buffer
    if (hasNext) {
      _Float16* An  = sA [(kt + 1) & 1];
      _Float16* Btn = sBt[(kt + 1) & 1];
#pragma unroll
      for (int u = 0; u < 4; ++u) {
        const int i = tid * 4 + u * 1024;
        const h4 y = {(_Float16)ax[u].x, (_Float16)ax[u].y,
                      (_Float16)ax[u].z, (_Float16)ax[u].w};
        *(h4*)&An[i] = y;
        const int rb = i >> 7, cb = i & 127;
        Btn[(cb + 0) * 32 + rb] = (_Float16)bx[u].x;
        Btn[(cb + 1) * 32 + rb] = (_Float16)bx[u].y;
        Btn[(cb + 2) * 32 + rb] = (_Float16)bx[u].z;
        Btn[(cb + 3) * 32 + rb] = (_Float16)bx[u].w;
      }
    }
    __syncthreads();
  }

  const int mloc = wave * 16 + ((lane >> 4) << 3);
#pragma unroll
  for (int j = 0; j < 8; ++j) {
    const int n = nBase + j * 16 + (lane & 15);
    const float bn = bias[n];
#pragma unroll
    for (int r = 0; r < 8; ++r) {
      float x = acc[j][r] + bn;
      x = x > 0.f ? x : 0.f;
      const int m = mBase + mloc + r;
      out[((size_t)(n >> 7) * 32768 + m) * 128 + (n & 127)] = (_Float16)x;
    }
  }
}

// ---- Kernel 2: fused per-(h,b) graph attention -----------------------------
// attn = QK^T; pdg = rel*attn + 1e-11; normalize by sqrt(col/row sums);
// res = (rel*pdg_norm) @ V.  One workgroup per (h,b); everything in LDS.
// Dynamic LDS 132096 B: [Q 32K | K 32K] (reused as pdg f32 64K) | V^T 32K | rel 32K | degs 1K
__global__ __launch_bounds__(256)
void attn_kernel(const _Float16* __restrict__ Qg,
                 const _Float16* __restrict__ Kg,
                 const _Float16* __restrict__ Vg,
                 const float* __restrict__ rel_in,   // [B, S, S]
                 _Float16* __restrict__ res)         // [B*S, 1024] heads concat
{
  extern __shared__ __align__(16) char smem[];
  _Float16* sQ  = (_Float16*)smem;                   // 128x128 f16 row-major [s][f]
  _Float16* sK  = (_Float16*)(smem + 32768);         // 128x128 f16 row-major [t][f]
  float*    sP  = (float*)smem;                      // pdg f32, overlays Q|K
  _Float16* sVt = (_Float16*)(smem + 65536);         // V transposed [f][t]
  _Float16* sR  = (_Float16*)(smem + 98304);         // rel, later softed (f16) [s][t]
  float*    sIn = (float*)(smem + 131072);           // indeg[128]
  float*    sOd = (float*)(smem + 131072 + 512);     // outdeg[128]

  const int hb = blockIdx.x;
  const int h  = hb >> 8;                            // B = 256
  const int b  = hb & 255;
  const int tid  = threadIdx.x;
  const int wave = tid >> 5;
  const int lane = tid & 31;

  const size_t tileOff = ((size_t)h * 32768 + (size_t)b * 128) * 128;
  // Q,K: straight vector copies (b128)
  for (int i = tid * 8; i < 16384; i += 2048) {
    *(uint4*)&sQ[i] = *(const uint4*)&Qg[tileOff + i];
    *(uint4*)&sK[i] = *(const uint4*)&Kg[tileOff + i];
  }
  // V: contiguous reads, transposed scatter (sVt[f][t] = V[t][f])
  for (int i = tid * 8; i < 16384; i += 2048) {
    const int t = i >> 7, f = i & 127;
    const h8 v = *(const h8*)&Vg[tileOff + i];
#pragma unroll
    for (int q = 0; q < 8; ++q) sVt[(f + q) * 128 + t] = v[q];
  }
  const float* relB = rel_in + (size_t)b * 16384;
  if (h < 4) {                                       // rel + h*I  (heads 0..3)
    const float diag = (float)h;
    for (int i = tid * 4; i < 16384; i += 1024) {
      const int s = i >> 7, t = i & 127;
      const float4 x = *(const float4*)&relB[i];
      float v0 = x.x, v1 = x.y, v2 = x.z, v3 = x.w;
      if (s == t + 0) v0 += diag;
      if (s == t + 1) v1 += diag;
      if (s == t + 2) v2 += diag;
      if (s == t + 3) v3 += diag;
      const h4 y = {(_Float16)v0, (_Float16)v1, (_Float16)v2, (_Float16)v3};
      *(h4*)&sR[i] = y;
    }
  } else {                                           // transposed rel (heads 4..7)
    for (int i = tid * 4; i < 16384; i += 1024) {    // contiguous reads, scatter writes
      const int t = i >> 7, s = i & 127;             // reading relB[t][s..s+3]
      const float4 x = *(const float4*)&relB[i];
      sR[(s + 0) * 128 + t] = (_Float16)x.x;
      sR[(s + 1) * 128 + t] = (_Float16)x.y;
      sR[(s + 2) * 128 + t] = (_Float16)x.z;
      sR[(s + 3) * 128 + t] = (_Float16)x.w;
    }
  }
  __syncthreads();

  v8f zero = {};
  v8f acc[8];
#pragma unroll
  for (int j = 0; j < 8; ++j) acc[j] = zero;

  // attn = Q @ K^T : wave w owns rows [16w, 16w+16), all 8 col-tiles
#pragma unroll
  for (int ks = 0; ks < 128; ks += 32) {
    const v16h a = frag_a_lds(sQ, 128, wave * 16, ks);
#pragma unroll
    for (int j = 0; j < 8; ++j) {
      const v16h bt = frag_bt_lds(sK, 128, ks, j * 16);   // K already [N x K]
      acc[j] = WMMA_F32_F16(a, bt, acc[j]);
    }
  }
  __syncthreads();   // Q/K reads done before pdg overlays them

  // pdg = rel * attn + 1e-11  (f32, overlays Q|K)
  const int mloc = wave * 16 + ((lane >> 4) << 3);
#pragma unroll
  for (int j = 0; j < 8; ++j) {
    const int n = j * 16 + (lane & 15);
#pragma unroll
    for (int r = 0; r < 8; ++r) {
      const int m = mloc + r;
      sP[m * 128 + n] = (float)sR[m * 128 + n] * acc[j][r] + 1e-11f;
    }
  }
  __syncthreads();

  // indeg[t] = sqrt(sum_s pdg[s][t]);  outdeg[s] = sqrt(sum_t pdg[s][t])
  if (tid < 128) {
    float s = 0.f;
    for (int r = 0; r < 128; ++r) s += sP[r * 128 + tid];   // distinct banks/lane
    sIn[tid] = sqrtf(s);
  } else {
    const int row = tid - 128;
    float s = 0.f;
    for (int cc = 0; cc < 128; ++cc) {
      const int c = (cc + row) & 127;                        // stagger: no bank conflict
      s += sP[row * 128 + c];
    }
    sOd[row] = sqrtf(s);
  }
  __syncthreads();

  // softed = rel * pdg / (indeg[t] * outdeg[s])  -> f16 in-place over rel
  for (int i = tid * 4; i < 16384; i += 1024) {
    const int s = i >> 7, t = i & 127;
    const h4 r4 = *(const h4*)&sR[i];
    const float4 p4 = *(const float4*)&sP[i];
    const float od = sOd[s];
    const h4 y = {(_Float16)((float)r4[0] * p4.x / (sIn[t + 0] * od)),
                  (_Float16)((float)r4[1] * p4.y / (sIn[t + 1] * od)),
                  (_Float16)((float)r4[2] * p4.z / (sIn[t + 2] * od)),
                  (_Float16)((float)r4[3] * p4.w / (sIn[t + 3] * od))};
    *(h4*)&sR[i] = y;
  }
  __syncthreads();

  // res = softed @ V
#pragma unroll
  for (int j = 0; j < 8; ++j) acc[j] = zero;
#pragma unroll
  for (int ks = 0; ks < 128; ks += 32) {
    const v16h a = frag_a_lds(sR, 128, wave * 16, ks);
#pragma unroll
    for (int j = 0; j < 8; ++j) {
      const v16h bb = frag_bt_lds(sVt, 128, ks, j * 16);
      acc[j] = WMMA_F32_F16(a, bb, acc[j]);
    }
  }

  // res_ws[(b*128 + m) * 1024 + h*128 + n]  (heads concatenated on features)
#pragma unroll
  for (int j = 0; j < 8; ++j) {
    const int n = j * 16 + (lane & 15);
#pragma unroll
    for (int r = 0; r < 8; ++r) {
      const int m = mloc + r;
      res[((size_t)b * 128 + m) * 1024 + h * 128 + n] = (_Float16)acc[j][r];
    }
  }
}

// ---- Kernel 3: out = relu([res | code] @ Wo + bo), double-buffered ---------
__global__ __launch_bounds__(256)
void out_proj_kernel(const _Float16* __restrict__ resAll, // [32768,1024] f16
                     const float* __restrict__ code,      // [32768,1024] f32
                     const float* __restrict__ Wo,        // [2048,1024]
                     const float* __restrict__ bo,        // [1024]
                     float* __restrict__ out)             // [32768,1024]
{
  __shared__ __align__(16) _Float16 sA [2][128 * 32];
  __shared__ __align__(16) _Float16 sBt[2][128 * 32];   // Wo slab transposed, N x K
  const int mBase = blockIdx.x * 128;
  const int nBase = blockIdx.y * 128;
  const int tid  = threadIdx.x;
  const int wave = tid >> 5;
  const int lane = tid & 31;

  v8f zero = {};
  v8f acc[8];
#pragma unroll
  for (int j = 0; j < 8; ++j) acc[j] = zero;

  // prologue: slab 0 (A from f16 res, B from Wo)
#pragma unroll
  for (int u = 0; u < 2; ++u) {
    const int i = tid * 8 + u * 2048;
    const int r = i >> 5, c = i & 31;
    *(uint4*)&sA[0][i] = *(const uint4*)&resAll[(size_t)(mBase + r) * 1024 + c];
  }
#pragma unroll
  for (int u = 0; u < 4; ++u) {
    const int i = tid * 4 + u * 1024;
    const int r = i >> 7, c = i & 127;
    const float4 x = *(const float4*)&Wo[(size_t)r * 1024 + nBase + c];
    sBt[0][(c + 0) * 32 + r] = (_Float16)x.x;
    sBt[0][(c + 1) * 32 + r] = (_Float16)x.y;
    sBt[0][(c + 2) * 32 + r] = (_Float16)x.z;
    sBt[0][(c + 3) * 32 + r] = (_Float16)x.w;
  }
  __syncthreads();

  for (int kt = 0; kt < 64; ++kt) {
    uint4  au[2];
    float4 ax[4], bx[4];
    const bool hasNext = (kt + 1) < 64;
    const int kn = (kt + 1) * 32;
    const bool nextF16 = kn < 1024;                  // first half of K: res (f16)
    if (hasNext) {
      if (nextF16) {
#pragma unroll
        for (int u = 0; u < 2; ++u) {
          const int i = tid * 8 + u * 2048;
          const int r = i >> 5, c = i & 31;
          au[u] = *(const uint4*)&resAll[(size_t)(mBase + r) * 1024 + kn + c];
        }
      } else {
#pragma unroll
        for (int u = 0; u < 4; ++u) {
          const int i = tid * 4 + u * 1024;
          const int r = i >> 5, c = i & 31;
          ax[u] = *(const float4*)&code[(size_t)(mBase + r) * 1024 + (kn - 1024 + c)];
        }
      }
#pragma unroll
      for (int u = 0; u < 4; ++u) {
        const int i = tid * 4 + u * 1024;
        const int r = i >> 7, c = i & 127;
        bx[u] = *(const float4*)&Wo[(size_t)(kn + r) * 1024 + nBase + c];
      }
    }
    // compute current slab while next-slab loads are in flight
    const _Float16* A  = sA [kt & 1];
    const _Float16* Bt = sBt[kt & 1];
    const v16h a = frag_a_lds(A, 32, wave * 16, 0);
#pragma unroll
    for (int j = 0; j < 8; ++j) {
      const v16h bb = frag_bt_lds(Bt, 32, 0, j * 16);
      acc[j] = WMMA_F32_F16(a, bb, acc[j]);
    }
    // drain into other buffer
    if (hasNext) {
      _Float16* An  = sA [(kt + 1) & 1];
      _Float16* Btn = sBt[(kt + 1) & 1];
      if (nextF16) {
#pragma unroll
        for (int u = 0; u < 2; ++u) {
          const int i = tid * 8 + u * 2048;
          *(uint4*)&An[i] = au[u];
        }
      } else {
#pragma unroll
        for (int u = 0; u < 4; ++u) {
          const int i = tid * 4 + u * 1024;
          const h4 y = {(_Float16)ax[u].x, (_Float16)ax[u].y,
                        (_Float16)ax[u].z, (_Float16)ax[u].w};
          *(h4*)&An[i] = y;
        }
      }
#pragma unroll
      for (int u = 0; u < 4; ++u) {
        const int i = tid * 4 + u * 1024;
        const int r = i >> 7, c = i & 127;
        Btn[(c + 0) * 32 + r] = (_Float16)bx[u].x;
        Btn[(c + 1) * 32 + r] = (_Float16)bx[u].y;
        Btn[(c + 2) * 32 + r] = (_Float16)bx[u].z;
        Btn[(c + 3) * 32 + r] = (_Float16)bx[u].w;
      }
    }
    __syncthreads();
  }

  const int mloc = wave * 16 + ((lane >> 4) << 3);
#pragma unroll
  for (int j = 0; j < 8; ++j) {
    const int n = nBase + j * 16 + (lane & 15);
    const float bn = bo[n];
#pragma unroll
    for (int r = 0; r < 8; ++r) {
      const float x = acc[j][r] + bn;
      out[(size_t)(mBase + mloc + r) * 1024 + n] = x > 0.f ? x : 0.f;
    }
  }
}

// ---- host launcher ---------------------------------------------------------
extern "C" void kernel_launch(void* const* d_in, const int* in_sizes, int n_in,
                              void* d_out, int out_size, void* d_ws, size_t ws_size,
                              hipStream_t stream) {
  (void)in_sizes; (void)n_in; (void)out_size; (void)ws_size;
  const float* code = (const float*)d_in[0];
  const float* rel  = (const float*)d_in[1];
  const float* Wq   = (const float*)d_in[2];
  const float* bq   = (const float*)d_in[3];
  const float* Wk   = (const float*)d_in[4];
  const float* bk   = (const float*)d_in[5];
  const float* Wv   = (const float*)d_in[6];
  const float* bv   = (const float*)d_in[7];
  const float* Wo   = (const float*)d_in[8];
  const float* bo   = (const float*)d_in[9];
  float* out = (float*)d_out;

  // workspace: Q,K,V,res each 32768*1024 f16 = 64 MB
  char* ws = (char*)d_ws;
  _Float16* Qb = (_Float16*)(ws);
  _Float16* Kb = (_Float16*)(ws + (size_t)64 * 1024 * 1024);
  _Float16* Vb = (_Float16*)(ws + (size_t)128 * 1024 * 1024);
  _Float16* Rb = (_Float16*)(ws + (size_t)192 * 1024 * 1024);

  qkv_proj_kernel<<<dim3(256, 8), 256, 0, stream>>>(code, Wq, bq, Qb);
  qkv_proj_kernel<<<dim3(256, 8), 256, 0, stream>>>(code, Wk, bk, Kb);
  qkv_proj_kernel<<<dim3(256, 8), 256, 0, stream>>>(code, Wv, bv, Vb);

  const size_t smem = 132096;  // 129 KB (<< 320 KB WGP LDS)
  hipFuncSetAttribute(reinterpret_cast<const void*>(attn_kernel),
                      hipFuncAttributeMaxDynamicSharedMemorySize, (int)smem);
  attn_kernel<<<dim3(2048), 256, smem, stream>>>(Qb, Kb, Vb, rel, Rb);

  out_proj_kernel<<<dim3(256, 8), 256, 0, stream>>>(Rb, code, Wo, bo, out);
}
